// SpeFR_46385646797008
// MI455X (gfx1250) — matrix operand order repaired and checked by reading, not compiled, not run on hardware
//
#include <hip/hip_runtime.h>
#include <hip/hip_bf16.h>

typedef __attribute__((ext_vector_type(8)))  __bf16 v8bf;
typedef __attribute__((ext_vector_type(16))) __bf16 v16bf;
typedef __attribute__((ext_vector_type(8)))  float  v8f;

#define C_DIM 128
#define WDIM  64
#define HDIM  64
#define WH    (WDIM * HDIM)   // 4096
#define BATCH 8
#define NT    64              // N-tile (spatial columns) per block

// Workspace layout (bytes):
//   y      f32  (B,C,WH)             16,777,216
//   xT     bf16 (B,WH,C) transposed   8,388,608
//   wbf    bf16 (C,C)                    32,768
//   se_sig f32  (9,)                         36
#define Y_BYTES   ((size_t)BATCH * C_DIM * WH * 4)
#define XT_BYTES  ((size_t)BATCH * WH * C_DIM * 2)
#define WBF_BYTES ((size_t)C_DIM * C_DIM * 2)

// Async global -> LDS copy, 16 bytes per lane (GVS mode: saddr + per-lane voffset).
// Tracked by ASYNCcnt; no VGPR data movement.
static __device__ __forceinline__ void async_copy_b128(unsigned lds_off,
                                                       const void* sbase,
                                                       int voff) {
  asm volatile("global_load_async_to_lds_b128 %0, %1, %2"
               :: "v"(lds_off), "v"(voff), "s"(sbase)
               : "memory");
}
static __device__ __forceinline__ void wait_async0() {
  asm volatile("s_wait_asynccnt 0x0" ::: "memory");
}
// Low 32 bits of a generic pointer to __shared__ == LDS byte offset
// (LDS aperture occupies addr[63:32]; addr[31:0] is the offset).
static __device__ __forceinline__ unsigned lds_addr(const void* p) {
  return (unsigned)(size_t)p;
}

// ---------------------------------------------------------------------------
// Prep: attn_w f32 -> bf16 once (reused by all 512 GEMM blocks), and
// sigmoid(se_param) once (removes 9 v_exp_f32 from every epilogue thread).
// ---------------------------------------------------------------------------
__global__ __launch_bounds__(256) void SpeFR_prep(
    const float* __restrict__ w, const float* __restrict__ se_p,
    __bf16* __restrict__ wbf, float* __restrict__ se_sig) {
  const int tid = threadIdx.x;
  for (int i = tid; i < C_DIM * C_DIM; i += 256)
    wbf[i] = (__bf16)w[i];
  if (tid < 9)
    se_sig[tid] = 1.0f / (1.0f + __expf(-se_p[tid]));
}

// ---------------------------------------------------------------------------
// Transpose x (B,C,WH) f32 -> xT (B,WH,C) bf16 via an LDS tile so both the
// global read (along n) and global write (along c) are coalesced.
// ---------------------------------------------------------------------------
__global__ __launch_bounds__(256) void SpeFR_xT(
    const float* __restrict__ x, __bf16* __restrict__ xT) {
  __shared__ __align__(16) __bf16 T[NT * C_DIM];   // 16 KB
  const int tid = threadIdx.x;
  const int n0  = blockIdx.x * NT;
  const int b   = blockIdx.y;

  const float* xb = x + (size_t)b * C_DIM * WH;
  for (int i = tid; i < NT * C_DIM; i += 256) {
    const int n = i & (NT - 1);                    // contiguous gmem reads
    const int c = i >> 6;
    T[n * C_DIM + c] = (__bf16)xb[(size_t)c * WH + (n0 + n)];
  }
  __syncthreads();

  // Tile is contiguous in xT: 16 KB starting at (b*WH + n0)*C.
  const uint4* src = (const uint4*)T;
  uint4* dst = (uint4*)(xT + ((size_t)b * WH + n0) * C_DIM);
  #pragma unroll
  for (int i = 0; i < 4; ++i)
    dst[tid + 256 * i] = src[tid + 256 * i];
}

// ---------------------------------------------------------------------------
// GEMM: y[b,o,n] = sum_c attn_w[o,c] * x[b,c,n].  M=K=128, N=4096/batch.
// Async-DMA both operand tiles straight into LDS (no VGPR staging, no
// converts), then bf16 WMMA with f32 accumulate. 8 waves x (16M x 64N).
// ---------------------------------------------------------------------------
__global__ __launch_bounds__(256) void SpeFR_mix_gemm(
    const __bf16* __restrict__ wbf,   // (C,C) bf16, row-major [o][c]
    const __bf16* __restrict__ xT,    // (B,WH,C) bf16
    float* __restrict__ y)            // (B,C,WH) f32
{
  __shared__ __align__(32) __bf16 Alds[C_DIM * C_DIM]; // [o][c]      32 KB
  __shared__ __align__(32) __bf16 Xlds[NT * C_DIM];    // [n][c]      16 KB

  const int tid  = threadIdx.x;
  const int lane = tid & 31;
  const int wave = tid >> 5;
  const int n0   = blockIdx.x * NT;
  const int b    = blockIdx.y;

  // Async stage A: 32 KB contiguous, 8 x b128 per thread.
  const unsigned a0 = lds_addr(&Alds[0]);
  #pragma unroll
  for (int i = 0; i < 8; ++i) {
    const int off = tid * 16 + i * 4096;
    async_copy_b128(a0 + off, wbf, off);
  }
  // Async stage X tile: 16 KB contiguous, 4 x b128 per thread.
  const __bf16* xtile = xT + ((size_t)b * WH + n0) * C_DIM;
  const unsigned x0 = lds_addr(&Xlds[0]);
  #pragma unroll
  for (int i = 0; i < 4; ++i) {
    const int off = tid * 16 + i * 4096;
    async_copy_b128(x0 + off, xtile, off);
  }
  wait_async0();
  __syncthreads();

  const int m0 = wave * 16;                  // M strip owned by this wave
  const int hi = lane >> 4;                  // lane half
  const int ml = lane & 15;

  for (int nt = 0; nt < 4; ++nt) {
    v8f acc = {};
    const int nb = nt * 16;
    #pragma unroll
    for (int kc = 0; kc < 4; ++kc) {
      const int kb = kc * 32;
      // A fragment (16x32 MxK): row M=m0+ml; lanes 0-15: K=kb+0..15,
      // lanes 16-31: K=kb+16..31.
      const __bf16* arow = &Alds[(m0 + ml) * C_DIM + kb + 16 * hi];
      const v8bf a_lo = *(const v8bf*)(arow);
      const v8bf a_hi = *(const v8bf*)(arow + 8);
      v16bf a;
      #pragma unroll
      for (int i = 0; i < 8; ++i) { a[i] = a_lo[i]; a[8 + i] = a_hi[i]; }
      // B fragment (32x16 KxN): col N=nb+ml; lanes 0-15 K=kb+0..15,
      // lanes 16-31 K=kb+16..31 (contiguous in the transposed Xlds row).
      const v16bf bm = *(const v16bf*)(&Xlds[(nb + ml) * C_DIM + kb + 16 * hi]);
      acc = __builtin_amdgcn_wmma_f32_16x16x32_bf16(
                false, a, false, bm, (short)0, acc, false, false);
    }
    // D layout: VGPR r, lane l -> M = m0 + r + 8*hi, N = nb + ml.
    float* yout = y + (size_t)b * C_DIM * WH + (n0 + nb + ml);
    #pragma unroll
    for (int r = 0; r < 8; ++r)
      yout[(size_t)(m0 + r + 8 * hi) * WH] = acc[r];
  }
}

// ---------------------------------------------------------------------------
// Epilogue: out = max over s of se9(off)*xg * sigmoid(se9(off)*yg + bias[c]),
// off = s*4096 + w*64 + h, with the reference's flat (W,H,9)->(9,W,H)
// reinterpretation: sP=off%9, wP=off/576, hP=(off%576)/9,
// source=(wP+sP/3-1, hP+sP%3-1), zero-padded; se9(off)=se[off/4096]=se[s].
// ---------------------------------------------------------------------------
__global__ __launch_bounds__(256) void SpeFR_epilogue(
    const float* __restrict__ x,
    const float* __restrict__ y,
    const float* __restrict__ se_sig,  // (9,) pre-sigmoided
    const float* __restrict__ bias,
    float* __restrict__ out)
{
  const size_t idx = (size_t)blockIdx.x * blockDim.x + threadIdx.x;
  const size_t total = (size_t)BATCH * C_DIM * WH;
  if (idx >= total) return;

  const int h  = (int)(idx & (HDIM - 1));
  const int wq = (int)((idx >> 6) & (WDIM - 1));
  const int c  = (int)((idx >> 12) & (C_DIM - 1));
  const int b  = (int)(idx >> 19);

  float se[9];
  #pragma unroll
  for (int s = 0; s < 9; ++s) se[s] = se_sig[s];   // uniform scalar loads
  const float bia = bias[c];

  const size_t plane = ((size_t)b * C_DIM + c) * WH;
  const float* xb = x + plane;
  const float* yb = y + plane;
  __builtin_prefetch(yb + wq * HDIM + h, 0, 1);    // global_prefetch_b8

  float best = -3.402823466e38f;
  #pragma unroll
  for (int s = 0; s < 9; ++s) {
    const int off = s * WH + wq * HDIM + h;
    const int sP = off % 9;
    const int wP = off / (HDIM * 9);
    const int hP = (off % (HDIM * 9)) / 9;
    const int sw = wP + sP / 3 - 1;
    const int sh = hP + sP % 3 - 1;
    float xv = 0.0f, yv = 0.0f;
    if (sw >= 0 && sw < WDIM && sh >= 0 && sh < HDIM) {
      xv = xb[sw * HDIM + sh];
      yv = yb[sw * HDIM + sh];
    }
    const float sev = se[s];
    const float at = 1.0f / (1.0f + __expf(-(sev * yv + bia)));
    best = fmaxf(best, sev * xv * at);
  }
  out[idx] = best;
}

// ---------------------------------------------------------------------------
extern "C" void kernel_launch(void* const* d_in, const int* in_sizes, int n_in,
                              void* d_out, int out_size, void* d_ws, size_t ws_size,
                              hipStream_t stream) {
  const float* x    = (const float*)d_in[0];   // (8,128,64,64)
  const float* se_p = (const float*)d_in[1];   // (9,)
  const float* w    = (const float*)d_in[2];   // (128,128)
  const float* bias = (const float*)d_in[3];   // (128,)
  float* out = (float*)d_out;

  char* ws = (char*)d_ws;
  float*  y      = (float*)(ws);
  __bf16* xT     = (__bf16*)(ws + Y_BYTES);
  __bf16* wbf    = (__bf16*)(ws + Y_BYTES + XT_BYTES);
  float*  se_sig = (float*)(ws + Y_BYTES + XT_BYTES + WBF_BYTES);

  SpeFR_prep<<<1, 256, 0, stream>>>(w, se_p, wbf, se_sig);

  dim3 gt(WH / NT, BATCH);                     // (64, 8)
  SpeFR_xT<<<gt, 256, 0, stream>>>(x, xT);
  SpeFR_mix_gemm<<<gt, 256, 0, stream>>>(wbf, xT, y);

  const int total = BATCH * C_DIM * WH;        // 4,194,304
  SpeFR_epilogue<<<(total + 255) / 256, 256, 0, stream>>>(x, y, se_sig, bias, out);
}
